// QuantLinear_15934328668265
// MI455X (gfx1250) — matrix operand order repaired and checked by reading, not compiled, not run on hardware
//
#include <hip/hip_runtime.h>

// ---------------------------------------------------------------------------
// GPTQ int4 linear (4096x4096, group=128) + rank-16 LoRA, fused for gfx1250.
// Dequant int4 -> f16 via packed-f16 bit tricks, v_wmma_f32_16x16x32_f16 core,
// double-buffered software pipeline (1 barrier / K-chunk), LoRA epilogue.
// ---------------------------------------------------------------------------

typedef _Float16 v16h __attribute__((ext_vector_type(16)));
typedef _Float16 v8h  __attribute__((ext_vector_type(8)));
typedef _Float16 v4h  __attribute__((ext_vector_type(4)));
typedef _Float16 v2h  __attribute__((ext_vector_type(2)));
typedef float    v8f  __attribute__((ext_vector_type(8)));
typedef float    v4f  __attribute__((ext_vector_type(4)));
typedef int      v4i  __attribute__((ext_vector_type(4)));

#define IN_F    4096
#define OUT_F   4096
#define TOKENS  8192
#define R_RANK  16
#define SCALING 2.0f

#define BM 128
#define BN 128
#define BK 64

// ---------------------------------------------------------------------------
// Pre-pass: T[TOKENS][16] = x @ lora_A^T  (tiny, memory-bound)
// ---------------------------------------------------------------------------
__global__ __launch_bounds__(256) void lora_xA_kernel(
    const float* __restrict__ x, const float* __restrict__ lora_A,
    float* __restrict__ T)
{
    const int row = blockIdx.x * 16 + (threadIdx.x >> 4);
    const int r   = threadIdx.x & 15;
    const float* xp = x + (size_t)row * IN_F;
    const float* ap = lora_A + (size_t)r * IN_F;
    float acc = 0.0f;
    for (int k = 0; k < IN_F; k += 4) {
        v4f xv = *(const v4f*)(xp + k);
        v4f av = *(const v4f*)(ap + k);
        acc += xv.x * av.x + xv.y * av.y + xv.z * av.z + xv.w * av.w;
    }
    T[(size_t)row * R_RANK + r] = acc;
}

// ---------------------------------------------------------------------------
// Main fused kernel: 128x128 tile / block, 8 waves (2x4), 64x32 per wave.
// ---------------------------------------------------------------------------
__global__ __launch_bounds__(256) void gptq_lora_gemm(
    const float*    __restrict__ x,
    const int*      __restrict__ qweight,   // [IN_F/8][OUT_F], 8x int4 along K
    const unsigned* __restrict__ qzeros,    // [N_GROUPS][OUT_F/8]
    const float*    __restrict__ scales,    // [N_GROUPS][OUT_F]
    const float*    __restrict__ loraB,     // [OUT_F][16]
    const float*    __restrict__ T,         // [TOKENS][16] = x @ A^T
    float*          __restrict__ out)
{
    __shared__ __align__(32) _Float16 Xs[2][BM][BK];    // K-major   (2x16 KB)
    __shared__ __align__(32) _Float16 Ws[2][BN][BK];    // N-major!  (2x16 KB)
    __shared__ __align__(16) float    Ts[BM][R_RANK];   // 8 KB
    __shared__ __align__(16) float    Bs[BN][R_RANK];   // 8 KB

    const int tid  = threadIdx.x;
    const int lane = tid & 31;
    const int wid  = tid >> 5;
    const int gm   = blockIdx.y * BM;
    const int gn   = blockIdx.x * BN;
    const int wm0  = (wid >> 2) * 64;   // wave row base (2 waves in M)
    const int wn0  = (wid & 3) * 32;    // wave col base (4 waves in N)

    v8f acc[4][2];
#pragma unroll
    for (int i = 0; i < 4; ++i)
#pragma unroll
        for (int j = 0; j < 2; ++j)
            acc[i][j] = (v8f)0.0f;

    // staging work split
    const int xrow16 = tid >> 4;        // 0..15 : row within a 16-row pass
    const int xk4    = (tid & 15) * 4;  // 0..60 : float4 k-offset
    const int qr     = tid >> 5;        // 0..7  : packed-int row (K = kc + 8*qr + b)
    const int qn4    = (tid & 31) * 4;  // 0..124: 4 consecutive output columns
    const int zsh    = (qn4 & 7) * 4;   // 0 or 16

    // register staging for the software pipeline
    v4f      rx[8];
    v4i      rq;
    v4f      rsc;
    unsigned rzz;

    // ---- global loads for one K-chunk into registers ----
    auto load_globals = [&](int kc) {
#pragma unroll
        for (int p = 0; p < 8; ++p)
            rx[p] = *(const v4f*)&x[(size_t)(gm + p * 16 + xrow16) * IN_F + kc + xk4];
        const int g = kc >> 7;  // group index (GROUP=128, BK=64)
        rq  = *(const v4i*)&qweight[(size_t)((kc >> 3) + qr) * OUT_F + gn + qn4];
        rsc = *(const v4f*)&scales[(size_t)g * OUT_F + gn + qn4];
        rzz = qzeros[(size_t)g * (OUT_F / 8) + ((gn + qn4) >> 3)];
    };

    // ---- convert/dequant registers -> LDS buffer ----
    auto store_stage = [&](int buf) {
#pragma unroll
        for (int p = 0; p < 8; ++p) {
            v4h hv = { (_Float16)rx[p].x, (_Float16)rx[p].y,
                       (_Float16)rx[p].z, (_Float16)rx[p].w };
            *(v4h*)&Xs[buf][p * 16 + xrow16][xk4] = hv;
        }
        // Packed-f16 GPTQ dequant: (q | 0x6400) as f16 == 1024 + q exactly;
        // (1024+q) - (1024+z) is an exact integer subtract in f16 (no
        // cancellation), then one packed multiply by the scale.
#pragma unroll
        for (int j = 0; j < 4; ++j) {
            const unsigned qq = (unsigned)rq[j];
            const unsigned z  = ((rzz >> (zsh + 4 * j)) & 0xFu) + 1u;
            const _Float16 sh = (_Float16)rsc[j];
            const _Float16 zb = (_Float16)(float)(1024u + z);
            unsigned u0 = ( qq        & 0x000F000Fu) | 0x64006400u;  // (q0,q4)
            unsigned u1 = ((qq >> 4)  & 0x000F000Fu) | 0x64006400u;  // (q1,q5)
            unsigned u2 = ((qq >> 8)  & 0x000F000Fu) | 0x64006400u;  // (q2,q6)
            unsigned u3 = ((qq >> 12) & 0x000F000Fu) | 0x64006400u;  // (q3,q7)
            v2h h0 = __builtin_bit_cast(v2h, u0);
            v2h h1 = __builtin_bit_cast(v2h, u1);
            v2h h2 = __builtin_bit_cast(v2h, u2);
            v2h h3 = __builtin_bit_cast(v2h, u3);
            v8h hv = { h0.x, h1.x, h2.x, h3.x, h0.y, h1.y, h2.y, h3.y };
            v8h zv = { zb, zb, zb, zb, zb, zb, zb, zb };
            v8h sv = { sh, sh, sh, sh, sh, sh, sh, sh };
            v8h wv = (hv - zv) * sv;                 // v_pk_add/mul_f16
            *(v8h*)&Ws[buf][qn4 + j][qr * 8] = wv;   // ds_store_b128
        }
    };

    // ---- WMMA compute over one LDS buffer ----
    auto compute = [&](int buf) {
#pragma unroll
        for (int ks = 0; ks < BK; ks += 32) {
            v16h bf[2];
#pragma unroll
            for (int j = 0; j < 2; ++j) {
                // B layout: lanes 0-15 hold K ks..ks+15, lanes 16-31 ks+16..+31
                const int bc = wn0 + 16 * j + (lane & 15);
                const int bk = ks + ((lane >> 4) << 4);
                bf[j] = *(const v16h*)&Ws[buf][bc][bk];  // 32B contiguous
            }
#pragma unroll
            for (int i = 0; i < 4; ++i) {
                // A layout: lanes 0-15 hold K ks..ks+15, lanes 16-31 ks+16..+31
                const int ar = wm0 + 16 * i + (lane & 15);
                const int ak = ks + ((lane >> 4) << 4);
                v8h alo = *(const v8h*)&Xs[buf][ar][ak];
                v8h ahi = *(const v8h*)&Xs[buf][ar][ak + 8];
                v16h af = __builtin_shufflevector(alo, ahi,
                    0, 1, 2, 3, 4, 5, 6, 7, 8, 9, 10, 11, 12, 13, 14, 15);
#pragma unroll
                for (int j = 0; j < 2; ++j) {
                    acc[i][j] = __builtin_amdgcn_wmma_f32_16x16x32_f16(
                        false, af, false, bf[j], (short)0, acc[i][j], false, false);
                }
            }
        }
    };

    // ---- software pipeline: prologue ----
    load_globals(0);
    store_stage(0);
    __syncthreads();

    // ---- main loop: 1 barrier per chunk; next chunk's global loads are
    //      issued before compute so their latency hides under the WMMAs ----
    for (int kc = 0; kc < IN_F; kc += BK) {
        const int  buf      = (kc >> 6) & 1;
        const bool has_next = (kc + BK) < IN_F;
        if (has_next) {
            load_globals(kc + BK);
            if (kc + 2 * BK < IN_F)  // warm L2/WGP$ one more chunk out
                __builtin_prefetch(&qweight[(size_t)(((kc + 2 * BK) >> 3) + qr) * OUT_F + gn + qn4], 0, 0);
        }
        compute(buf);
        if (has_next)
            store_stage(buf ^ 1);
        __syncthreads();
    }

    // ---- stage LoRA tiles: Ts = T[gm..gm+127], Bs = loraB[gn..gn+127] ----
    {
        const int r = tid >> 1;
        const int c = (tid & 1) * 8;
        *(v4f*)&Ts[r][c]     = *(const v4f*)&T[(size_t)(gm + r) * R_RANK + c];
        *(v4f*)&Ts[r][c + 4] = *(const v4f*)&T[(size_t)(gm + r) * R_RANK + c + 4];
        *(v4f*)&Bs[r][c]     = *(const v4f*)&loraB[(size_t)(gn + r) * R_RANK + c];
        *(v4f*)&Bs[r][c + 4] = *(const v4f*)&loraB[(size_t)(gn + r) * R_RANK + c + 4];
    }
    __syncthreads();

    // ---- epilogue: out = acc + 2 * (Ts . Bs), per C/D VGPR layout ----
    // VGPR jj of a 16x16 f32 tile: lanes 0-15 -> M=jj, lanes 16-31 -> M=jj+8
#pragma unroll
    for (int i = 0; i < 4; ++i) {
#pragma unroll
        for (int j = 0; j < 2; ++j) {
            const int nl = wn0 + 16 * j + (lane & 15);
            const int n  = gn + nl;
            v4f b0 = *(const v4f*)&Bs[nl][0];
            v4f b1 = *(const v4f*)&Bs[nl][4];
            v4f b2 = *(const v4f*)&Bs[nl][8];
            v4f b3 = *(const v4f*)&Bs[nl][12];
#pragma unroll
            for (int jj = 0; jj < 8; ++jj) {
                const int ml = wm0 + 16 * i + jj + ((lane >> 4) << 3);
                v4f t0 = *(const v4f*)&Ts[ml][0];
                v4f t1 = *(const v4f*)&Ts[ml][4];
                v4f t2 = *(const v4f*)&Ts[ml][8];
                v4f t3 = *(const v4f*)&Ts[ml][12];
                float lora = t0.x * b0.x + t0.y * b0.y + t0.z * b0.z + t0.w * b0.w
                           + t1.x * b1.x + t1.y * b1.y + t1.z * b1.z + t1.w * b1.w
                           + t2.x * b2.x + t2.y * b2.y + t2.z * b2.z + t2.w * b2.w
                           + t3.x * b3.x + t3.y * b3.y + t3.z * b3.z + t3.w * b3.w;
                out[(size_t)(gm + ml) * OUT_F + n] = acc[i][j][jj] + SCALING * lora;
            }
        }
    }
}

// ---------------------------------------------------------------------------
// Launch
// ---------------------------------------------------------------------------
extern "C" void kernel_launch(void* const* d_in, const int* in_sizes, int n_in,
                              void* d_out, int out_size, void* d_ws, size_t ws_size,
                              hipStream_t stream) {
    const float*    x       = (const float*)d_in[0];
    const int*      qweight = (const int*)d_in[1];
    const unsigned* qzeros  = (const unsigned*)d_in[2];
    const float*    scales  = (const float*)d_in[3];
    /* d_in[4] = g_idx: arange(IN_F)//128, computed analytically in-kernel */
    const float*    lora_A  = (const float*)d_in[5];
    const float*    lora_B  = (const float*)d_in[6];
    float*          out     = (float*)d_out;
    float*          T       = (float*)d_ws;   // TOKENS * 16 floats = 512 KB scratch

    lora_xA_kernel<<<dim3(TOKENS / 16), dim3(256), 0, stream>>>(x, lora_A, T);
    gptq_lora_gemm<<<dim3(OUT_F / BN, TOKENS / BM), dim3(256), 0, stream>>>(
        x, qweight, qzeros, scales, lora_B, T, out);
}
